// DTIPredictor_17051020165713
// MI455X (gfx1250) — matrix operand order repaired and checked by reading, not compiled, not run on hardware
//
#include <hip/hip_runtime.h>

// ---------------------------------------------------------------------------
// DTI predictor, MI455X (gfx1250, wave32).
// bf16 operands + f32 accumulate WMMA GEMMs; bf16 projected tables to halve
// gather traffic; async global->LDS staging; folded 11->128 edge projections.
// ---------------------------------------------------------------------------

#define NLn 100000
#define NPn 200000
#define En  500000
#define Bn  256
#define Hh  128

typedef __bf16 bf16;
typedef __bf16 v16bf __attribute__((ext_vector_type(16)));
typedef __bf16 v8bf  __attribute__((ext_vector_type(8)));
typedef __bf16 v4bf  __attribute__((ext_vector_type(4)));
typedef float  v8f   __attribute__((ext_vector_type(8)));
typedef float  v4f   __attribute__((ext_vector_type(4)));

// ---- CDNA5 async global->LDS copy (16B per lane-op, ASYNCcnt tracked) ------
__device__ __forceinline__ void async_copy16(unsigned lds_off, const void* gaddr) {
  asm volatile("global_load_async_to_lds_b128 %0, %1, off"
               :: "v"(lds_off), "v"(gaddr) : "memory");
}
__device__ __forceinline__ void async_wait0() {
  asm volatile("s_wait_asynccnt 0" ::: "memory");
}

// ---- helpers ---------------------------------------------------------------
__device__ __forceinline__ unsigned fenc(float f) {
  unsigned u = __float_as_uint(f);
  return (u & 0x80000000u) ? ~u : (u | 0x80000000u);
}
__device__ __forceinline__ float fdec(unsigned u) {
  return __uint_as_float((u & 0x80000000u) ? (u & 0x7fffffffu) : ~u);
}

// Convert f32 weight [K x 128] to bf16 TRANSPOSED [128 x K] so WMMA B
// fragments become two contiguous 16B runs in LDS.
__global__ void cvt_t_bf16(const float* __restrict__ s, bf16* __restrict__ d, int K) {
  int i = blockIdx.x * blockDim.x + threadIdx.x;
  if (i < K * 128) { int k = i >> 7, c = i & 127; d[c * K + k] = (bf16)s[i]; }
}

__global__ void fill_f32(float* p, float v, int n) {
  int i = blockIdx.x * blockDim.x + threadIdx.x;
  if (i < n) p[i] = v;
}
__global__ void fill_u32(unsigned* p, unsigned v, int n) {
  int i = blockIdx.x * blockDim.x + threadIdx.x;
  if (i < n) p[i] = v;
}

// Fold lin_edge (11->128) + 128->128 projection into one 11->128 affine map.
__global__ void fold_edge(const float* __restrict__ Wl, const float* __restrict__ bl,
                          const float* __restrict__ P, const float* __restrict__ pb,
                          float* __restrict__ Mo, float* __restrict__ co) {
  int j = threadIdx.x;  // 128
  float c = pb[j];
  for (int k = 0; k < 128; ++k) c += bl[k] * P[k * 128 + j];
  co[j] = c;
  for (int r = 0; r < 11; ++r) {
    float m = 0.f;
    for (int k = 0; k < 128; ++k) m += Wl[r * 128 + k] * P[k * 128 + j];
    Mo[r * 128 + j] = m;
  }
}

// ---- WMMA GEMM: out[N x 128](bf16) = A[N x K] @ W[K x 128] + bias ----------
// Block: 256 thr (8 waves); tile 64 rows x 128 cols. Wave w owns col-tile w
// and reuses its B fragment across 4 row-tiles -> 4 WMMAs per K-step.
// Wt is the bf16 weight stored TRANSPOSED [128 x K].

template <int K, int AF32>
__global__ __launch_bounds__(256)
void gemm_bf16_k(const void* __restrict__ A, const bf16* __restrict__ Wt,
                 const float* __restrict__ bias, bf16* __restrict__ out, int N) {
  constexpr int LDA = 136;  // pad: 272B row stride = 68 dwords -> bank spread
  __shared__ bf16 lA[64 * LDA];
  __shared__ bf16 lWt[128 * LDA];
  const int t = threadIdx.x;
  const int row0 = blockIdx.x * 64;
  const unsigned lAo = (unsigned)(uintptr_t)(void*)&lA[0];
  const unsigned lWo = (unsigned)(uintptr_t)(void*)&lWt[0];

  // Stage transposed weights (async DMA, 16B chunks; hot in L2 across blocks).
  for (int ch = t; ch < 128 * (K / 8); ch += 256) {
    int c = ch / (K / 8), k8 = ch % (K / 8);
    async_copy16(lWo + (unsigned)(c * LDA + k8 * 8) * 2, Wt + (size_t)ch * 8);
  }
  // Stage A tile.
  if (AF32) {
    const float* Af = (const float*)A;
    for (int idx = t; idx < 64 * K; idx += 256) {
      int r = idx / K, c = idx - r * K;
      int gr = row0 + r;
      lA[r * LDA + c] = (gr < N) ? (bf16)Af[(size_t)gr * K + c] : (bf16)0.0f;
    }
  } else {
    const bf16* Ab = (const bf16*)A;
    for (int ch = t; ch < 64 * (K / 8); ch += 256) {
      int r = ch / (K / 8), k8 = ch % (K / 8);
      int gr = row0 + r;
      if (gr < N) {
        async_copy16(lAo + (unsigned)(r * LDA + k8 * 8) * 2,
                     Ab + (size_t)gr * K + (size_t)k8 * 8);
      } else {
        bf16* p = &lA[r * LDA + k8 * 8];
#pragma unroll
        for (int j = 0; j < 8; ++j) p[j] = (bf16)0.0f;
      }
    }
  }
  async_wait0();
  __syncthreads();

  const int lane = t & 31;
  const int colb = (t >> 5) * 16;
  const int row = lane & 15;
  const int ka = (lane >> 4) * 8;   // A fragment k-offset (branch-free layout)
  const int kb = (lane >> 4) * 8;   // B fragment k-offset
  const int col = colb + row;

  v8f acc[4] = {};
#pragma unroll
  for (int k0 = 0; k0 < K; k0 += 32) {
    v8bf b0 = *(const v8bf*)&lWt[col * LDA + k0 + kb];
    v8bf b1 = *(const v8bf*)&lWt[col * LDA + k0 + kb + 16];
    v16bf bfv = __builtin_shufflevector(b0, b1, 0, 1, 2, 3, 4, 5, 6, 7,
                                        8, 9, 10, 11, 12, 13, 14, 15);
#pragma unroll
    for (int rb = 0; rb < 4; ++rb) {
      const bf16* ar = &lA[(rb * 16 + row) * LDA + k0];
      v8bf a0 = *(const v8bf*)(ar + ka);
      v8bf a1 = *(const v8bf*)(ar + 16 + ka);
      v16bf af = __builtin_shufflevector(a0, a1, 0, 1, 2, 3, 4, 5, 6, 7,
                                         8, 9, 10, 11, 12, 13, 14, 15);
      acc[rb] = __builtin_amdgcn_wmma_f32_16x16x32_bf16(false, af, false, bfv,
                                                        (short)0, acc[rb],
                                                        false, false);
    }
  }

  // Epilogue: bias + bf16 store. Full-tile fast path (only the last block of
  // a launch can be partial) keeps the hot path branch-free.
  const float bv = bias[col];
  const int rbase = (lane < 16) ? 0 : 8;
  bf16* po = out + (size_t)(row0 + rbase) * 128 + col;
  if (row0 + 64 <= N) {
#pragma unroll
    for (int rb = 0; rb < 4; ++rb) {
#pragma unroll
      for (int r = 0; r < 8; ++r)
        po[(size_t)(rb * 16 + r) * 128] = (bf16)(acc[rb][r] + bv);
    }
  } else {
#pragma unroll
    for (int rb = 0; rb < 4; ++rb) {
#pragma unroll
      for (int r = 0; r < 8; ++r) {
        int gr = row0 + rb * 16 + rbase + r;
        if (gr < N) out[(size_t)gr * 128 + col] = (bf16)(acc[rb][r] + bv);
      }
    }
  }
}

// ---- Edge pass 1: atom logits + attention logits + per-dst max -------------
__global__ __launch_bounds__(256)
void edge_pass1(const float* __restrict__ e, const int* __restrict__ src,
                const int* __restrict__ dst, const int* __restrict__ gid,
                const float* __restrict__ MI, const float* __restrict__ cI,
                const float* __restrict__ MP, const float* __restrict__ cP,
                const bf16* __restrict__ tAs, const bf16* __restrict__ tAd,
                const bf16* __restrict__ tPs, const bf16* __restrict__ tPd,
                const float* __restrict__ fcw, const float* __restrict__ fcb1,
                const float* __restrict__ attw, const float* __restrict__ attb1,
                const float* __restrict__ alpha1,
                float* __restrict__ w_att, unsigned* __restrict__ mdst,
                float* __restrict__ ap, int E, int epb) {
  const int wave = threadIdx.x >> 5, lane = threadIdx.x & 31;
  const int c0 = lane * 4;
  const int pw = epb >> 3;
  int start = blockIdx.x * epb + wave * pw;
  int end = start + pw; if (end > E) end = E;

  const float fcb = fcb1[0], attb = attb1[0], alpha = alpha1[0];
  const v4f fw = *(const v4f*)(fcw + c0);
  const v4f aw = *(const v4f*)(attw + c0);
  const v4f cI4 = *(const v4f*)(cI + c0);
  const v4f cP4 = *(const v4f*)(cP + c0);

  float acc = 0.f; int curg = -1;
  for (int i = start; i < end; ++i) {
    const int s_ = src[i], d_ = dst[i], g_ = gid[i];
    v4f Ie = cI4, Pe = cP4;
    for (int k = 0; k < 11; ++k) {
      const float ek = e[(size_t)i * 11 + k];
      Ie += ek * (*(const v4f*)(MI + k * 128 + c0));
      Pe += ek * (*(const v4f*)(MP + k * 128 + c0));
    }
    const size_t so = (size_t)s_ * 128 + c0, dof = (size_t)d_ * 128 + c0;
    const v4bf as4 = *(const v4bf*)(tAs + so);
    const v4bf ad4 = *(const v4bf*)(tAd + dof);
    const v4bf ps4 = *(const v4bf*)(tPs + so);
    const v4bf pd4 = *(const v4bf*)(tPd + dof);
    float p1 = 0.f, p2 = 0.f;
#pragma unroll
    for (int j = 0; j < 4; ++j) {
      p1 += Ie[j] * (float)as4[j] * (float)ad4[j] * fw[j];
      const float tt = (float)ps4[j] + (float)pd4[j] + Pe[j];
      p2 += (tt >= 0.f ? tt : alpha * tt) * aw[j];
    }
#pragma unroll
    for (int m = 16; m > 0; m >>= 1) {
      p1 += __shfl_xor(p1, m, 32);
      p2 += __shfl_xor(p2, m, 32);
    }
    if (lane == 0) {
      const float w = p2 + attb;
      w_att[i] = w;
      atomicMax(&mdst[d_], fenc(w));
      if (g_ != curg) {
        if (curg >= 0) atomicAdd(&ap[curg], acc);
        curg = g_; acc = 0.f;
      }
      acc += p1 + fcb;
    }
  }
  if (lane == 0 && curg >= 0) atomicAdd(&ap[curg], acc);
}

// ---- Edge pass 2: exp(w - max) and per-dst denominator ---------------------
__global__ void edge_pass2(const float* __restrict__ w_att, const int* __restrict__ dst,
                           const unsigned* __restrict__ mdst,
                           float* __restrict__ ew, float* __restrict__ z, int E) {
  int i = blockIdx.x * blockDim.x + threadIdx.x;
  if (i >= E) return;
  const int d_ = dst[i];
  const float v = expf(w_att[i] - fdec(mdst[d_]));
  ew[i] = v;
  atomicAdd(&z[d_], v);
}

// ---- Edge pass 3: a * We * Ws[src] * Wd[dst] segment-summed into s[B,128] --
__global__ __launch_bounds__(256)
void edge_pass3(const float* __restrict__ e, const int* __restrict__ src,
                const int* __restrict__ dst, const int* __restrict__ gid,
                const float* __restrict__ MW, const float* __restrict__ cW,
                const bf16* __restrict__ tWs, const bf16* __restrict__ tWd,
                const float* __restrict__ ew, const float* __restrict__ z,
                float* __restrict__ sacc, int E, int epb) {
  const int wave = threadIdx.x >> 5, lane = threadIdx.x & 31;
  const int c0 = lane * 4;
  const int pw = epb >> 3;
  int start = blockIdx.x * epb + wave * pw;
  int end = start + pw; if (end > E) end = E;

  const v4f cW4 = *(const v4f*)(cW + c0);
  v4f acc = {};
  int curg = -1;
  for (int i = start; i < end; ++i) {
    const int s_ = src[i], d_ = dst[i], g_ = gid[i];
    const float a = ew[i] / z[d_];
    v4f We = cW4;
    for (int k = 0; k < 11; ++k) {
      const float ek = e[(size_t)i * 11 + k];
      We += ek * (*(const v4f*)(MW + k * 128 + c0));
    }
    if (g_ != curg) {
      if (curg >= 0) {
#pragma unroll
        for (int j = 0; j < 4; ++j) atomicAdd(&sacc[(size_t)curg * 128 + c0 + j], acc[j]);
      }
      curg = g_;
      acc = (v4f){};
    }
    const size_t so = (size_t)s_ * 128 + c0, dof = (size_t)d_ * 128 + c0;
    const v4bf ws4 = *(const v4bf*)(tWs + so);
    const v4bf wd4 = *(const v4bf*)(tWd + dof);
#pragma unroll
    for (int j = 0; j < 4; ++j) acc[j] += a * We[j] * (float)ws4[j] * (float)wd4[j];
  }
  if (curg >= 0) {
#pragma unroll
    for (int j = 0; j < 4; ++j) atomicAdd(&sacc[(size_t)curg * 128 + c0 + j], acc[j]);
  }
}

// ---- FC head: 128 -> 200 -> 200 -> 1 (LReLU + eval-mode BN) ----------------
#define BN_SC 0.99999500003749937f  // 1/sqrt(1 + 1e-5)

__global__ void head_l1(const float* __restrict__ s, const float* __restrict__ w,
                        const float* __restrict__ b, const float* __restrict__ g,
                        const float* __restrict__ bb, float* __restrict__ x1) {
  const int bi = blockIdx.x, j = threadIdx.x;
  if (j >= 200) return;
  float acc = b[j];
  for (int k = 0; k < 128; ++k) acc += s[bi * 128 + k] * w[k * 200 + j];
  acc = acc >= 0.f ? acc : 0.01f * acc;
  x1[bi * 200 + j] = acc * BN_SC * g[j] + bb[j];
}
__global__ void head_l2(const float* __restrict__ x1, const float* __restrict__ w,
                        const float* __restrict__ b, const float* __restrict__ g,
                        const float* __restrict__ bb, float* __restrict__ x2) {
  const int bi = blockIdx.x, j = threadIdx.x;
  if (j >= 200) return;
  float acc = b[j];
  for (int k = 0; k < 200; ++k) acc += x1[bi * 200 + k] * w[k * 200 + j];
  acc = acc >= 0.f ? acc : 0.01f * acc;
  x2[bi * 200 + j] = acc * BN_SC * g[j] + bb[j];
}
__global__ void head_out(const float* __restrict__ x2, const float* __restrict__ w,
                         const float* __restrict__ b1, const float* __restrict__ ap,
                         float* __restrict__ out) {
  const int bi = blockIdx.x * blockDim.x + threadIdx.x;
  if (bi >= Bn) return;
  float acc = b1[0];
  for (int k = 0; k < 200; ++k) acc += x2[bi * 200 + k] * w[k];
  out[bi] = ap[bi] - acc;
}

// ---------------------------------------------------------------------------
// Host launch. Input index map: setup_inputs() dict insertion order, params
// flattened depth-first in source order ({w,b} per linear layer).
// ---------------------------------------------------------------------------
extern "C" void kernel_launch(void* const* d_in, const int* in_sizes, int n_in,
                              void* d_out, int out_size, void* d_ws, size_t ws_size,
                              hipStream_t stream) {
  (void)in_sizes; (void)n_in; (void)out_size; (void)ws_size;
  auto F = [&](int i) { return (const float*)d_in[i]; };
  auto I = [&](int i) { return (const int*)d_in[i]; };

  char* base = (char*)d_ws;
  size_t off = 0;
  auto alloc = [&](size_t bytes) -> void* {
    void* p = base + off;
    off = (off + bytes + 255) & ~(size_t)255;
    return p;
  };

  bf16* hl = (bf16*)alloc((size_t)NLn * Hh * 2);
  bf16* hp = (bf16*)alloc((size_t)NPn * Hh * 2);
  bf16* wnl = (bf16*)alloc(64 * Hh * 2);   // transposed [128 x 64]
  bf16* wnp = (bf16*)alloc(64 * Hh * 2);
  // per branch: [atom prj_src, atom prj_dst, bias prj_src, bias prj_dst, bias w_src, bias w_dst]
  const int projIdx[12] = {18, 20, 34, 36, 40, 42, 24, 26, 59, 61, 65, 67};
  bf16* wp[12];
  for (int i = 0; i < 12; ++i) wp[i] = (bf16*)alloc(Hh * Hh * 2);  // transposed
  float *foldM[6], *foldC[6];
  for (int i = 0; i < 6; ++i) {
    foldM[i] = (float*)alloc(11 * Hh * 4);
    foldC[i] = (float*)alloc(Hh * 4);
  }
  char* tb = (char*)alloc((size_t)3 * (NLn + NPn) * Hh * 2);
  float* w_att = (float*)alloc((size_t)En * 4);
  float* ewb   = (float*)alloc((size_t)En * 4);
  unsigned* mEnc = (unsigned*)alloc((size_t)NPn * 4);
  float* zb  = (float*)alloc((size_t)NPn * 4);
  float* sb  = (float*)alloc((size_t)Bn * Hh * 4);
  float* apb = (float*)alloc(Bn * 4);
  float* x1  = (float*)alloc(Bn * 200 * 4);
  float* x2  = (float*)alloc(Bn * 200 * 4);

  // Weight conversions (bf16 + transpose, tiny).
  cvt_t_bf16<<<(64 * Hh + 255) / 256, 256, 0, stream>>>(F(10), wnl, 64);
  cvt_t_bf16<<<(64 * Hh + 255) / 256, 256, 0, stream>>>(F(12), wnp, 64);
  for (int i = 0; i < 12; ++i)
    cvt_t_bf16<<<(Hh * Hh + 255) / 256, 256, 0, stream>>>(F(projIdx[i]), wp[i], 128);

  // Folded edge projections: [lp: I,P,W][pl: I,P,W]
  const int foldW[6][2] = {{14, 22}, {14, 38}, {14, 44}, {16, 28}, {16, 63}, {16, 69}};
  for (int i = 0; i < 6; ++i)
    fold_edge<<<1, 128, 0, stream>>>(F(foldW[i][0]), F(foldW[i][0] + 1),
                                     F(foldW[i][1]), F(foldW[i][1] + 1),
                                     foldM[i], foldC[i]);

  // Node embeddings (WMMA, K=64, f32 A inputs).
  gemm_bf16_k<64, 1><<<(NLn + 63) / 64, 256, 0, stream>>>(F(0), wnl, F(11), hl, NLn);
  gemm_bf16_k<64, 1><<<(NPn + 63) / 64, 256, 0, stream>>>(F(1), wnp, F(13), hp, NPn);

  const int epb = 1024;
  const int gb = (En + epb - 1) / epb;

  for (int br = 0; br < 2; ++br) {
    const int Ns = br ? NPn : NLn, Nd = br ? NLn : NPn;
    bf16* hs = br ? hp : hl;
    bf16* hd = br ? hl : hp;
    const float* ee = F(br ? 3 : 2);
    const int* srcI = I(br ? 7 : 4);
    const int* dstI = I(br ? 8 : 5);
    const int* gidI = I(br ? 9 : 6);
    const int wB  = br ? 6 : 0;
    const int ib  = br ? 59 : 34;  // bias_* dict base (prj_src.w)
    const int aSb = br ? 25 : 19;  // atom prj_src.b
    const int aDb = br ? 27 : 21;  // atom prj_dst.b
    const int fcwI = br ? 32 : 30, fcbI = br ? 33 : 31;

    const size_t es = (size_t)Hh * 2;
    bf16* tAs = (bf16*)(tb + 0 * (size_t)Ns * es);
    bf16* tPs = (bf16*)(tb + 1 * (size_t)Ns * es);
    bf16* tWs = (bf16*)(tb + 2 * (size_t)Ns * es);
    char* tdb = tb + 3 * (size_t)Ns * es;
    bf16* tAd = (bf16*)(tdb + 0 * (size_t)Nd * es);
    bf16* tPd = (bf16*)(tdb + 1 * (size_t)Nd * es);
    bf16* tWd = (bf16*)(tdb + 2 * (size_t)Nd * es);

    // Projected node tables (WMMA, K=128, bf16 A), bf16 outputs.
    gemm_bf16_k<128, 0><<<(Ns + 63) / 64, 256, 0, stream>>>(hs, wp[wB + 0], F(aSb),    tAs, Ns);
    gemm_bf16_k<128, 0><<<(Ns + 63) / 64, 256, 0, stream>>>(hs, wp[wB + 2], F(ib + 1), tPs, Ns);
    gemm_bf16_k<128, 0><<<(Ns + 63) / 64, 256, 0, stream>>>(hs, wp[wB + 4], F(ib + 7), tWs, Ns);
    gemm_bf16_k<128, 0><<<(Nd + 63) / 64, 256, 0, stream>>>(hd, wp[wB + 1], F(aDb),    tAd, Nd);
    gemm_bf16_k<128, 0><<<(Nd + 63) / 64, 256, 0, stream>>>(hd, wp[wB + 3], F(ib + 3), tPd, Nd);
    gemm_bf16_k<128, 0><<<(Nd + 63) / 64, 256, 0, stream>>>(hd, wp[wB + 5], F(ib + 9), tWd, Nd);

    fill_f32<<<1, 256, 0, stream>>>(apb, 0.f, Bn);
    fill_f32<<<(Bn * Hh + 255) / 256, 256, 0, stream>>>(sb, 0.f, Bn * Hh);
    fill_f32<<<(Nd + 255) / 256, 256, 0, stream>>>(zb, 0.f, Nd);
    fill_u32<<<(Nd + 255) / 256, 256, 0, stream>>>(mEnc, 0u, Nd);

    edge_pass1<<<gb, 256, 0, stream>>>(ee, srcI, dstI, gidI,
                                       foldM[br * 3 + 0], foldC[br * 3 + 0],
                                       foldM[br * 3 + 1], foldC[br * 3 + 1],
                                       tAs, tAd, tPs, tPd,
                                       F(fcwI), F(fcbI), F(ib + 13), F(ib + 14), F(ib + 12),
                                       w_att, mEnc, apb, En, epb);
    edge_pass2<<<(En + 255) / 256, 256, 0, stream>>>(w_att, dstI, mEnc, ewb, zb, En);
    edge_pass3<<<gb, 256, 0, stream>>>(ee, srcI, dstI, gidI,
                                       foldM[br * 3 + 2], foldC[br * 3 + 2],
                                       tWs, tWd, ewb, zb, sb, En, epb);

    head_l1<<<Bn, 256, 0, stream>>>(sb, F(ib + 15), F(ib + 16), F(ib + 17), F(ib + 18), x1);
    head_l2<<<Bn, 256, 0, stream>>>(x1, F(ib + 19), F(ib + 20), F(ib + 21), F(ib + 22), x2);
    head_out<<<1, 256, 0, stream>>>(x2, F(ib + 23), F(ib + 24), apb, (float*)d_out + br * Bn);
  }
}
